// SignNetLayer_84748294684980
// MI455X (gfx1250) — matrix operand order repaired and checked by reading, not compiled, not run on hardware
//
#include <hip/hip_runtime.h>
#include <hip/hip_bf16.h>
#include <cstdint>
#include <cstddef>

typedef _Float16 f16;
typedef __attribute__((ext_vector_type(16))) _Float16 v16h;
typedef __attribute__((ext_vector_type(8)))  float    v8f;

#define HID   64
#define LSTMH 96
#define NG    384   /* 4*LSTMH */
#define BN_BLOCKS 240

// ---------------------------------------------------------------------------
// Pack a K x No weight matrix (optionally transposed source) into the exact
// per-lane B-fragment layout of V_WMMA_F32_16X16X32_F16 (zero-padded to Kp).
// Fragment layout (wave32): lane l holds column n = nt*16 + (l&15) and 16
// K values: element e at kt*32 + (l>>4)*8 + e (e<8), kt*32 + 16 + (l>>4)*8 + e-8 (e>=8).
// ---------------------------------------------------------------------------
__global__ __launch_bounds__(256)
void pack_b_k(const float* __restrict__ B, f16* __restrict__ out,
              int K, int Kp, int No, int ldb, int trans)
{
  int gid = blockIdx.x * 256 + threadIdx.x;
  int total = Kp * No;
  if (gid >= total) return;
  int e    = gid & 15;
  int lane = (gid >> 4) & 31;
  int rest = gid >> 9;
  int ntiles = No >> 4;
  int nt = rest % ntiles;
  int kt = rest / ntiles;
  int k = kt * 32 + (e < 8 ? (lane >> 4) * 8 + e : 16 + (lane >> 4) * 8 + (e - 8));
  int n = nt * 16 + (lane & 15);
  float v = 0.0f;
  if (k < K) v = trans ? B[(size_t)n * ldb + k] : B[(size_t)k * ldb + n];
  out[gid] = (f16)v;
}

// ---------------------------------------------------------------------------
// Branch-free A-fragment loader: 4 aligned float4 loads + f32->f16 convert.
// Requires lda % 4 == 0 and K padded to multiples of 32.
// ---------------------------------------------------------------------------
template <bool HAS_A2>
__device__ __forceinline__ v16h load_a_frag(const float* __restrict__ A1,
                                            const float* __restrict__ A2,
                                            float a2c, int row, int lda, int kb,
                                            bool ok)
{
  v16h a;
  if (ok) {
    const float* ap = A1 + (size_t)row * lda + kb;
    float4 x0 = *(const float4*)(ap);
    float4 x1 = *(const float4*)(ap + 4);
    float4 x2 = *(const float4*)(ap + 16);
    float4 x3 = *(const float4*)(ap + 20);
    if (HAS_A2) {
      const float* ap2 = A2 + (size_t)row * lda + kb;
      float4 y0 = *(const float4*)(ap2);
      float4 y1 = *(const float4*)(ap2 + 4);
      float4 y2 = *(const float4*)(ap2 + 16);
      float4 y3 = *(const float4*)(ap2 + 20);
      x0.x += a2c * y0.x; x0.y += a2c * y0.y; x0.z += a2c * y0.z; x0.w += a2c * y0.w;
      x1.x += a2c * y1.x; x1.y += a2c * y1.y; x1.z += a2c * y1.z; x1.w += a2c * y1.w;
      x2.x += a2c * y2.x; x2.y += a2c * y2.y; x2.z += a2c * y2.z; x2.w += a2c * y2.w;
      x3.x += a2c * y3.x; x3.y += a2c * y3.y; x3.z += a2c * y3.z; x3.w += a2c * y3.w;
    }
    a[0]  = (f16)x0.x; a[1]  = (f16)x0.y; a[2]  = (f16)x0.z; a[3]  = (f16)x0.w;
    a[4]  = (f16)x1.x; a[5]  = (f16)x1.y; a[6]  = (f16)x1.z; a[7]  = (f16)x1.w;
    a[8]  = (f16)x2.x; a[9]  = (f16)x2.y; a[10] = (f16)x2.z; a[11] = (f16)x2.w;
    a[12] = (f16)x3.x; a[13] = (f16)x3.y; a[14] = (f16)x3.z; a[15] = (f16)x3.w;
  } else {
    #pragma unroll
    for (int j = 0; j < 16; j++) a[j] = (f16)0.0f;
  }
  return a;
}

// ---------------------------------------------------------------------------
// WMMA GEMM: D[M x No] = act( (A1 + a2c*A2) @ Bpacked + bias (+ D if ACCUM) )
// f32 A converted to f16 in-register, f32 accumulation.
// 256 threads = 8 waves; each wave computes a 32x64 tile (2 M-subtiles x 4
// N-subtiles = 8 accumulators; B fragments reused across both M-subtiles).
// K must be a multiple of 32 (activations zero-padded); KSTEPS = K/32.
// All epilogue behavior is compile-time specialized (no runtime branches).
// ---------------------------------------------------------------------------
template <int KSTEPS, bool HAS_A2, bool ACCUM, bool RELU, bool HAS_BIAS>
__global__ __launch_bounds__(256)
void gemm_wmma_k(int M, int No,
                 const float* __restrict__ A1,
                 const float* __restrict__ A2, float a2c,
                 const float* __restrict__ epsPtr, int lda,
                 const f16* __restrict__ Bp,
                 const float* __restrict__ bias,
                 float* __restrict__ D, int ldd)
{
  if (HAS_A2 && epsPtr) a2c += epsPtr[0];   // a2c = 1 + eps[i]
  const int lane = threadIdx.x & 31;
  const int wave = threadIdx.x >> 5;
  const int m0   = (blockIdx.x * 8 + wave) * 32;
  if (m0 >= M) return;                      // wave-uniform
  const int n0   = blockIdx.y * 64;
  const int half = lane >> 4;
  const int l15  = lane & 15;
  const int ntiles = No >> 4;

  const int row0 = m0 + l15;
  const int row1 = m0 + 16 + l15;
  const bool ok0 = row0 < M;                // uniform when M % 16 == 0
  const bool ok1 = row1 < M;

  v8f acc[2][4];
  #pragma unroll
  for (int mt = 0; mt < 2; mt++)
    #pragma unroll
    for (int nt = 0; nt < 4; nt++)
      #pragma unroll
      for (int r = 0; r < 8; r++) acc[mt][nt][r] = 0.0f;

  #pragma unroll
  for (int ks = 0; ks < KSTEPS; ks++) {
    const int kb = ks * 32 + half * 8;
    // Issue all loads for this k-step before the first WMMA.
    v16h a0 = load_a_frag<HAS_A2>(A1, A2, a2c, row0, lda, kb, ok0);
    v16h a1 = load_a_frag<HAS_A2>(A1, A2, a2c, row1, lda, kb, ok1);
    const f16* bbase = Bp + (((size_t)ks * ntiles + (n0 >> 4)) * 32 + lane) * 16;
    v16h b0 = *(const v16h*)(bbase);
    v16h b1 = *(const v16h*)(bbase + 32 * 16);
    v16h b2 = *(const v16h*)(bbase + 64 * 16);
    v16h b3 = *(const v16h*)(bbase + 96 * 16);
    acc[0][0] = __builtin_amdgcn_wmma_f32_16x16x32_f16(false, a0, false, b0, (short)0, acc[0][0], false, false);
    acc[1][0] = __builtin_amdgcn_wmma_f32_16x16x32_f16(false, a1, false, b0, (short)0, acc[1][0], false, false);
    acc[0][1] = __builtin_amdgcn_wmma_f32_16x16x32_f16(false, a0, false, b1, (short)0, acc[0][1], false, false);
    acc[1][1] = __builtin_amdgcn_wmma_f32_16x16x32_f16(false, a1, false, b1, (short)0, acc[1][1], false, false);
    acc[0][2] = __builtin_amdgcn_wmma_f32_16x16x32_f16(false, a0, false, b2, (short)0, acc[0][2], false, false);
    acc[1][2] = __builtin_amdgcn_wmma_f32_16x16x32_f16(false, a1, false, b2, (short)0, acc[1][2], false, false);
    acc[0][3] = __builtin_amdgcn_wmma_f32_16x16x32_f16(false, a0, false, b3, (short)0, acc[0][3], false, false);
    acc[1][3] = __builtin_amdgcn_wmma_f32_16x16x32_f16(false, a1, false, b3, (short)0, acc[1][3], false, false);
  }

  // Epilogue: D layout: VGPR r -> row mbase + 8*half + r, col = n0+nt*16+l15.
  #pragma unroll
  for (int mt = 0; mt < 2; mt++) {
    const int mbase = m0 + mt * 16;
    if (mbase >= M) break;                  // uniform
    const bool full = (mbase + 16) <= M;    // uniform fast path
    #pragma unroll
    for (int nt = 0; nt < 4; nt++) {
      const int col = n0 + nt * 16 + l15;
      const float bsum = HAS_BIAS ? bias[col] : 0.0f;
      float* dp = D + (size_t)(mbase + half * 8) * ldd + col;
      if (full) {
        if (ACCUM) {
          float cin[8];
          #pragma unroll
          for (int r = 0; r < 8; r++) cin[r] = dp[(size_t)r * ldd];
          #pragma unroll
          for (int r = 0; r < 8; r++) {
            float v = acc[mt][nt][r] + bsum + cin[r];
            if (RELU) v = fmaxf(v, 0.0f);
            dp[(size_t)r * ldd] = v;
          }
        } else {
          #pragma unroll
          for (int r = 0; r < 8; r++) {
            float v = acc[mt][nt][r] + bsum;
            if (RELU) v = fmaxf(v, 0.0f);
            dp[(size_t)r * ldd] = v;
          }
        }
      } else {
        #pragma unroll
        for (int r = 0; r < 8; r++) {
          const int rr = mbase + half * 8 + r;
          if (rr < M) {
            float v = acc[mt][nt][r] + bsum;
            if (ACCUM) v += dp[(size_t)r * ldd];
            if (RELU)  v = fmaxf(v, 0.0f);
            dp[(size_t)r * ldd] = v;
          }
        }
      }
    }
  }
}

// ---------------------------------------------------------------------------
// Fused edge kernel: e = edge_attr @ lin_e_w + lin_e_b (weights in LDS),
// m = relu(h[src] + e), atomic scatter-add into agg[dst].
// One thread handles one edge x 2 channels. hStride/aggStride allow the
// zero-padded N x 32 layer-0 layout.
// ---------------------------------------------------------------------------
__global__ __launch_bounds__(256)
void edge_msg_k(const float* __restrict__ h,
                const int* __restrict__ src, const int* __restrict__ dst,
                const float* __restrict__ eattr,
                const float* __restrict__ lew, const float* __restrict__ leb,
                float* __restrict__ agg, int E, int ic, int hStride, int aggStride)
{
  __shared__ float lw[16 * 64];
  __shared__ float lb[64];
  for (int i = threadIdx.x; i < 16 * ic; i += 256) lw[i] = lew[i];
  for (int i = threadIdx.x; i < ic;      i += 256) lb[i] = leb[i];
  __syncthreads();
  const int tpe = ic >> 1;
  int gid = blockIdx.x * 256 + threadIdx.x;
  int e = gid / tpe;
  if (e >= E) return;
  int c = (gid - e * tpe) * 2;
  int s = src[e], d = dst[e];
  float e0 = lb[c], e1 = lb[c + 1];
  const float* ap = eattr + (size_t)e * 16;
  #pragma unroll
  for (int k = 0; k < 16; k++) {
    float a = ap[k];
    e0 += a * lw[k * ic + c];
    e1 += a * lw[k * ic + c + 1];
  }
  const float* hp = h + (size_t)s * hStride + c;
  float m0 = hp[0] + e0; m0 = m0 > 0.0f ? m0 : 0.0f;
  float m1 = hp[1] + e1; m1 = m1 > 0.0f ? m1 : 0.0f;
  atomicAdd(agg + (size_t)d * aggStride + c,     m0);
  atomicAdd(agg + (size_t)d * aggStride + c + 1, m1);
}

// ---------------------------------------------------------------------------
// Deterministic two-stage BN statistics over the node axis (64 channels).
// ---------------------------------------------------------------------------
__global__ __launch_bounds__(256)
void bn_stats_k(const float* __restrict__ pre, float* __restrict__ part, int N)
{
  __shared__ float ss[256], sq[256];
  int c  = threadIdx.x & 63;
  int rs = threadIdx.x >> 6;
  float s = 0.0f, q = 0.0f;
  for (int i = blockIdx.x * 4 + rs; i < N; i += gridDim.x * 4) {
    float v = pre[(size_t)i * 64 + c];
    s += v; q += v * v;
  }
  ss[threadIdx.x] = s; sq[threadIdx.x] = q;
  __syncthreads();
  if (rs == 0) {
    #pragma unroll
    for (int k = 1; k < 4; k++) { s += ss[threadIdx.x + 64 * k]; q += sq[threadIdx.x + 64 * k]; }
    part[blockIdx.x * 128 + c]      = s;
    part[blockIdx.x * 128 + 64 + c] = q;
  }
}

__global__ void bn_final_k(const float* __restrict__ part, int G,
                           const float* __restrict__ gamma, const float* __restrict__ beta,
                           float* __restrict__ sc, float* __restrict__ sh, float invN)
{
  int c = threadIdx.x;
  if (c >= 64) return;
  float s = 0.0f, q = 0.0f;
  for (int g = 0; g < G; g++) { s += part[g * 128 + c]; q += part[g * 128 + 64 + c]; }
  float mean = s * invN;
  float var  = q * invN - mean * mean;     // biased variance (torch-style)
  float scl  = gamma[c] * rsqrtf(var + 1e-5f);
  sc[c] = scl;
  sh[c] = beta[c] - mean * scl;
}

__global__ __launch_bounds__(256)
void bn_apply_relu_k(const float* __restrict__ pre,
                     const float* __restrict__ sc, const float* __restrict__ sh,
                     float* __restrict__ out, int total)
{
  int idx = blockIdx.x * 256 + threadIdx.x;
  if (idx >= total) return;
  int c = idx & 63;
  float v = pre[idx] * sc[c] + sh[c];
  out[idx] = v > 0.0f ? v : 0.0f;
}

// h[n][c] = c < 16 ? s*x[n][c] : 0   (zero-padded N x 32 layer-0 input)
__global__ __launch_bounds__(256)
void sign_copy_pad_k(const float* __restrict__ x, float* __restrict__ h, float s, int N)
{
  int idx = blockIdx.x * 256 + threadIdx.x;
  if (idx >= N * 32) return;
  int n = idx >> 5, c = idx & 31;
  h[idx] = (c < 16) ? s * x[n * 16 + c] : 0.0f;
}

__global__ void bias_comb_k(const float* a, const float* b,
                            const float* c2, const float* d2,
                            float* o1, float* o2)
{
  int i = threadIdx.x;
  if (i < NG) { o1[i] = a[i] + b[i]; o2[i] = c2[i] + d2[i]; }
}

// ---------------------------------------------------------------------------
// LSTM cell elementwise: gates (i,f,g,o) from gbuf[N x 384], update c, write h.
// ---------------------------------------------------------------------------
__global__ __launch_bounds__(256)
void lstm_cell_k(const float* __restrict__ g, float* __restrict__ c,
                 float* __restrict__ hout, int N)
{
  int idx = blockIdx.x * 256 + threadIdx.x;
  if (idx >= N * LSTMH) return;
  int n = idx / LSTMH, j = idx - n * LSTMH;
  const float* gp = g + (size_t)n * NG;
  float gi = gp[j], gf = gp[LSTMH + j], gg = gp[2 * LSTMH + j], go = gp[3 * LSTMH + j];
  float si = 1.0f / (1.0f + expf(-gi));
  float sf = 1.0f / (1.0f + expf(-gf));
  float so = 1.0f / (1.0f + expf(-go));
  float cc = sf * c[idx] + si * tanhf(gg);
  c[idx]    = cc;
  hout[idx] = so * tanhf(cc);
}

// ---------------------------------------------------------------------------
// JK attention: one wave per node. alpha[t] = softmax_t(<[hf;hb], att_w>+b),
// jkout = sum_t alpha[t] * seq[t].
// ---------------------------------------------------------------------------
__global__ __launch_bounds__(256)
void attn_jk_k(const float* __restrict__ hf, const float* __restrict__ hb,
               const float* __restrict__ seq,
               const float* __restrict__ attw, const float* __restrict__ attb,
               float* __restrict__ jkout, int N)
{
  const int lane = threadIdx.x & 31;
  const int wave = threadIdx.x >> 5;
  const int node = blockIdx.x * 8 + wave;
  if (node >= N) return;
  float s[3];
  #pragma unroll
  for (int t = 0; t < 3; t++) {
    const float* pf = hf + ((size_t)t * N + node) * LSTMH;
    const float* pb = hb + ((size_t)t * N + node) * LSTMH;
    float p = 0.0f;
    #pragma unroll
    for (int j0 = 0; j0 < 3; j0++) {
      int j = lane * 3 + j0;
      p += pf[j] * attw[j] + pb[j] * attw[LSTMH + j];
    }
    #pragma unroll
    for (int m = 16; m >= 1; m >>= 1) p += __shfl_xor(p, m, 32);
    s[t] = p + attb[0];
  }
  float mx = fmaxf(s[0], fmaxf(s[1], s[2]));
  float e0 = expf(s[0] - mx), e1 = expf(s[1] - mx), e2 = expf(s[2] - mx);
  float inv = 1.0f / (e0 + e1 + e2);
  float a0 = e0 * inv, a1 = e1 * inv, a2 = e2 * inv;
  #pragma unroll
  for (int cc = 0; cc < 2; cc++) {
    int c = lane + cc * 32;
    size_t base = (size_t)node * HID + c;
    jkout[base] = a0 * seq[base] + a1 * seq[(size_t)N * HID + base]
                + a2 * seq[(size_t)2 * N * HID + base];
  }
}

// ---------------------------------------------------------------------------
extern "C" void kernel_launch(void* const* d_in, const int* in_sizes, int n_in,
                              void* d_out, int out_size, void* d_ws, size_t ws_size,
                              hipStream_t stream)
{
  const int N = in_sizes[0] / 16;     // IN_DIM = 16
  const int E = in_sizes[1] / 2;

  const float* x     = (const float*)d_in[0];
  const int*   srcI  = (const int*)d_in[1];
  const int*   dstI  = srcI + E;
  const float* eattr = (const float*)d_in[2];
  const float *lew[3], *leb[3], *w1[3], *b1[3], *gam[3], *bet[3], *w2[3], *b2[3];
  for (int i = 0; i < 3; i++) {
    const int base = 3 + i * 8;
    lew[i] = (const float*)d_in[base + 0];
    leb[i] = (const float*)d_in[base + 1];
    w1[i]  = (const float*)d_in[base + 2];
    b1[i]  = (const float*)d_in[base + 3];
    gam[i] = (const float*)d_in[base + 4];
    bet[i] = (const float*)d_in[base + 5];
    w2[i]  = (const float*)d_in[base + 6];
    b2[i]  = (const float*)d_in[base + 7];
  }
  const float* eps     = (const float*)d_in[27];
  const float* wi_f    = (const float*)d_in[28];
  const float* wh_f    = (const float*)d_in[29];
  const float* bi_f    = (const float*)d_in[30];
  const float* bh_f    = (const float*)d_in[31];
  const float* wi_b    = (const float*)d_in[32];
  const float* wh_b    = (const float*)d_in[33];
  const float* bi_b    = (const float*)d_in[34];
  const float* bh_b    = (const float*)d_in[35];
  const float* attw    = (const float*)d_in[36];
  const float* attb    = (const float*)d_in[37];
  const float* jkw     = (const float*)d_in[38];
  const float* jkb     = (const float*)d_in[39];
  const float* rho_w1  = (const float*)d_in[40];
  const float* rho_b1  = (const float*)d_in[41];
  const float* rho_g   = (const float*)d_in[42];
  const float* rho_be  = (const float*)d_in[43];
  const float* rho_w2  = (const float*)d_in[44];
  const float* rho_b2  = (const float*)d_in[45];
  float* out = (float*)d_out;

  // ---- workspace carve-out ------------------------------------------------
  char*  wsb = (char*)d_ws;
  size_t off = 0;
  auto alloc = [&](size_t bytes) -> char* {
    char* p = wsb + off;
    off = (off + bytes + 255) & ~(size_t)255;
    return p;
  };
  float* bufA   = (float*)alloc((size_t)N * 64 * 4);       // h0 (N x 32 padded)
  float* bufB   = (float*)alloc((size_t)N * 64 * 4);       // agg / BN-act
  float* bufC   = (float*)alloc((size_t)N * 64 * 4);       // pre-BN
  float* seq    = (float*)alloc((size_t)3 * N * 64 * 4);
  float* hf     = (float*)alloc((size_t)3 * N * 96 * 4);
  float* hb     = (float*)alloc((size_t)3 * N * 96 * 4);
  float* cst    = (float*)alloc((size_t)N * 96 * 4);
  float* gbuf   = (float*)alloc((size_t)N * 384 * 4);
  float* jkout  = (float*)alloc((size_t)N * 64 * 4);
  float* phiacc = (float*)alloc((size_t)N * 64 * 4);
  float* bnpart = (float*)alloc((size_t)BN_BLOCKS * 128 * 4);
  float* bnsc   = (float*)alloc(64 * 4);
  float* bnsh   = (float*)alloc(64 * 4);
  float* bcf    = (float*)alloc(NG * 4);
  float* bcb    = (float*)alloc(NG * 4);
  f16* pw1[3], *pw2[3];
  for (int i = 0; i < 3; i++) pw1[i] = (f16*)alloc(64 * 64 * 2);
  for (int i = 0; i < 3; i++) pw2[i] = (f16*)alloc(64 * 64 * 2);
  f16* pwi_f = (f16*)alloc((size_t)64 * 384 * 2);
  f16* pwh_f = (f16*)alloc((size_t)96 * 384 * 2);
  f16* pwi_b = (f16*)alloc((size_t)64 * 384 * 2);
  f16* pwh_b = (f16*)alloc((size_t)96 * 384 * 2);
  f16* pjk   = (f16*)alloc(64 * 64 * 2);
  f16* prho1 = (f16*)alloc(64 * 64 * 2);
  f16* prho2 = (f16*)alloc(64 * 64 * 2);
  if (off > ws_size) return;   // workspace too small: bail (avoids UB)

  // ---- helpers ------------------------------------------------------------
  auto pack = [&](const float* B, f16* o, int K, int No, int ldb, int trans) {
    int Kp = (K + 31) & ~31;
    int total = Kp * No;
    pack_b_k<<<(total + 255) / 256, 256, 0, stream>>>(B, o, K, Kp, No, ldb, trans);
  };
  // K must be a multiple of 32 here (activations padded where needed).
  // Dispatch to the compile-time specialization matching this network's uses.
  auto gemm = [&](int M, int K, int No, const float* A1, const float* A2,
                  const float* epsPtr, int lda, const f16* Bp,
                  const float* bias, float* D, int ldd, bool accum, bool relu) {
    dim3 g((M + 255) / 256, No / 64);
    const int ks = K / 32;
    if (A2) {                                  // conv matmul1 (fused agg+(1+eps)h)
      if (ks == 1)
        gemm_wmma_k<1, true, false, false, true><<<g, 256, 0, stream>>>(M, No, A1, A2, 1.0f, epsPtr, lda, Bp, bias, D, ldd);
      else
        gemm_wmma_k<2, true, false, false, true><<<g, 256, 0, stream>>>(M, No, A1, A2, 1.0f, epsPtr, lda, Bp, bias, D, ldd);
    } else if (accum && !bias) {               // LSTM gate2: gbuf += h@Wh^T (K=96)
      gemm_wmma_k<3, false, true, false, false><<<g, 256, 0, stream>>>(M, No, A1, nullptr, 0.0f, nullptr, lda, Bp, bias, D, ldd);
    } else if (accum) {                        // jk: phiacc += jkout@jk_w + jk_b
      gemm_wmma_k<2, false, true, false, true><<<g, 256, 0, stream>>>(M, No, A1, nullptr, 0.0f, nullptr, lda, Bp, bias, D, ldd);
    } else if (relu) {                         // conv matmul2 (fused ReLU)
      gemm_wmma_k<2, false, false, true, true><<<g, 256, 0, stream>>>(M, No, A1, nullptr, 0.0f, nullptr, lda, Bp, bias, D, ldd);
    } else {                                   // gate1 / rho matmuls
      gemm_wmma_k<2, false, false, false, true><<<g, 256, 0, stream>>>(M, No, A1, nullptr, 0.0f, nullptr, lda, Bp, bias, D, ldd);
    }
  };
  auto bn = [&](float* pre, const float* g_, const float* be_) {
    bn_stats_k<<<BN_BLOCKS, 256, 0, stream>>>(pre, bnpart, N);
    bn_final_k<<<1, 64, 0, stream>>>(bnpart, BN_BLOCKS, g_, be_, bnsc, bnsh, 1.0f / (float)N);
    bn_apply_relu_k<<<(N * 64 + 255) / 256, 256, 0, stream>>>(pre, bnsc, bnsh, bufB, N * 64);
  };

  // ---- pack all weights into WMMA B-fragment layout -----------------------
  pack(w1[0], pw1[0], 16, 64, 64, 0);   // zero-padded to Kp=32
  pack(w1[1], pw1[1], 64, 64, 64, 0);
  pack(w1[2], pw1[2], 64, 64, 64, 0);
  for (int i = 0; i < 3; i++) pack(w2[i], pw2[i], 64, 64, 64, 0);
  pack(wi_f, pwi_f, 64, NG, 64, 1);     // B = Wi^T  (Wi is [384 x 64])
  pack(wh_f, pwh_f, 96, NG, 96, 1);     // B = Wh^T  (Wh is [384 x 96])
  pack(wi_b, pwi_b, 64, NG, 64, 1);
  pack(wh_b, pwh_b, 96, NG, 96, 1);
  pack(jkw,    pjk,   64, 64, 64, 0);
  pack(rho_w1, prho1, 64, 64, 64, 0);
  pack(rho_w2, prho2, 64, 64, 64, 0);
  bias_comb_k<<<1, NG, 0, stream>>>(bi_f, bh_f, bi_b, bh_b, bcf, bcb);

  hipMemsetAsync(phiacc, 0, (size_t)N * 64 * 4, stream);

  // ---- phi(x) + phi(-x) ---------------------------------------------------
  for (int sgn = 0; sgn < 2; sgn++) {
    const float sv = sgn == 0 ? 1.0f : -1.0f;
    sign_copy_pad_k<<<(N * 32 + 255) / 256, 256, 0, stream>>>(x, bufA, sv, N);
    const float* hcur = bufA;

    for (int i = 0; i < 3; i++) {
      const int ic  = (i == 0) ? 16 : 64;
      const int str = (i == 0) ? 32 : 64;   // zero-padded stride (multiple of 32)
      hipMemsetAsync(bufB, 0, (size_t)N * str * 4, stream);
      {
        int total = E * (ic >> 1);
        edge_msg_k<<<(total + 255) / 256, 256, 0, stream>>>(
            hcur, srcI, dstI, eattr, lew[i], leb[i], bufB, E, ic, str, str);
      }
      // pre = (agg + (1+eps_i)*h) @ w1 + b1   (K = str, zero-padded for i==0)
      gemm(N, str, 64, bufB, hcur, eps + i, str, pw1[i], b1[i], bufC, 64, false, false);
      bn(bufC, gam[i], bet[i]);                    // bufB = relu(BN(pre))
      // seq[i] = relu(bufB @ w2 + b2)
      gemm(N, 64, 64, bufB, nullptr, nullptr, 64, pw2[i], b2[i],
           seq + (size_t)i * N * 64, 64, false, true);
      hcur = seq + (size_t)i * N * 64;
    }

    // bidirectional LSTM over the 3 layer outputs
    for (int dir = 0; dir < 2; dir++) {
      float*       hdir = dir == 0 ? hf : hb;
      const f16*   pwi  = dir == 0 ? pwi_f : pwi_b;
      const f16*   pwh  = dir == 0 ? pwh_f : pwh_b;
      const float* bc   = dir == 0 ? bcf : bcb;
      hipMemsetAsync(cst, 0, (size_t)N * 96 * 4, stream);
      for (int stp = 0; stp < 3; stp++) {
        int t = dir == 0 ? stp : 2 - stp;
        gemm(N, 64, NG, seq + (size_t)t * N * 64, nullptr, nullptr, 64,
             pwi, bc, gbuf, NG, false, false);
        if (stp > 0) {
          int tp = dir == 0 ? t - 1 : t + 1;
          gemm(N, 96, NG, hdir + (size_t)tp * N * 96, nullptr, nullptr, 96,
               pwh, nullptr, gbuf, NG, true, false);
        }
        lstm_cell_k<<<(N * 96 + 255) / 256, 256, 0, stream>>>(
            gbuf, cst, hdir + (size_t)t * N * 96, N);
      }
    }

    attn_jk_k<<<(N + 7) / 8, 256, 0, stream>>>(hf, hb, seq, attw, attb, jkout, N);
    // phiacc += jkout @ jk_w + jk_b
    gemm(N, 64, 64, jkout, nullptr, nullptr, 64, pjk, jkb, phiacc, 64, true, false);
  }

  // ---- rho ----------------------------------------------------------------
  gemm(N, 64, 64, phiacc, nullptr, nullptr, 64, prho1, rho_b1, bufC, 64, false, false);
  bn(bufC, rho_g, rho_be);                         // bufB = relu(BN(pre))
  gemm(N, 64, 64, bufB, nullptr, nullptr, 64, prho2, rho_b2, out, 64, false, false);
  (void)n_in; (void)out_size;
}
